// GraphConvolution_73856257622121
// MI455X (gfx1250) — matrix operand order repaired and checked by reading, not compiled, not run on hardware
//
#include <hip/hip_runtime.h>
#include <hip/hip_bf16.h>

// ---------- types matching CDNA5 WMMA builtin signatures ----------
typedef __attribute__((ext_vector_type(16))) __bf16 bf16x16;
typedef __attribute__((ext_vector_type(8)))  __bf16 bf16x8;
typedef __attribute__((ext_vector_type(8)))  float  f32x8;
typedef __attribute__((ext_vector_type(4)))  int    i32x4;

union BF16X16U { bf16x16 v; bf16x8 h[2]; };

// Problem sizes (fixed by the reference)
#define NN 4096
#define EE 8192
#define INV 512
#define OUTV 512
#define INE 128

#define TILE_K 32
#define LDT 40   // padded LDS leading dim (80 B rows -> 16B-aligned chunks)

#define GLOBAL_AS __attribute__((address_space(1)))
#define LDS_AS    __attribute__((address_space(3)))

// ---- CDNA5 async global->LDS copy (ASYNCcnt path), guarded w/ fallback ----
#if defined(__has_builtin)
# if __has_builtin(__builtin_amdgcn_global_load_async_to_lds_b128)
#  define USE_ASYNC 1
# endif
#endif
#ifndef USE_ASYNC
# define USE_ASYNC 0
#endif

#if USE_ASYNC
__device__ __forceinline__ void cp_async16(const __bf16* g, __bf16* l) {
  GLOBAL_AS i32x4* gp = (GLOBAL_AS i32x4*)(GLOBAL_AS void*)g;
  LDS_AS   i32x4* lp = (LDS_AS   i32x4*)(LDS_AS   void*)l;
  __builtin_amdgcn_global_load_async_to_lds_b128(gp, lp, 0, 0);
}
# if __has_builtin(__builtin_amdgcn_s_wait_asynccnt)
#  define WAIT_ASYNC(n) __builtin_amdgcn_s_wait_asynccnt(n)
# else
#  define WAIT_ASYNC(n) asm volatile("s_wait_asynccnt %0" ::"i"(n) : "memory")
# endif
#endif

// ---------------- fragment loaders (ISA 7.12.2 bf16 layouts) ----------------
// A (16x32, MxK): lanes 0-15 -> M=lane, lanes 16-31 -> M=lane-16.
// For lane-half h, element e: K = (e>>3)*16 + h*8 + (e&7)
// => two contiguous 8-bf16 (16B) chunks at K = h*8 and K = h*8+16.
__device__ __forceinline__ bf16x16 load_a_frag(const __bf16* tile, int mBase, int lane) {
  int h = (lane >> 4) & 1;
  int m = lane & 15;
  const __bf16* row = tile + (mBase + m) * LDT;
  BF16X16U r;
  r.h[0] = *(const bf16x8*)(row + h * 8);
  r.h[1] = *(const bf16x8*)(row + h * 8 + 16);
  return r.v;
}

// B (32x16, KxN): lane = N column (lanes 16-31 repeat N);
// per lane-half h: K chunks at h*8 and h*8+16 (same as A). Tile stored as [n][k].
__device__ __forceinline__ bf16x16 load_b_frag(const __bf16* tile, int nBase, int lane) {
  int h = (lane >> 4) & 1;
  int n = lane & 15;
  const __bf16* row = tile + (nBase + n) * LDT;
  BF16X16U r;
  r.h[0] = *(const bf16x8*)(row + h * 8);
  r.h[1] = *(const bf16x8*)(row + h * 8 + 16);
  return r.v;
}

__device__ __forceinline__ f32x8 wmma_bf16(bf16x16 a, bf16x16 b, f32x8 c) {
  return __builtin_amdgcn_wmma_f32_16x16x32_bf16(
      false, a, false, b, (short)0, c, false, false);
}

// ---------------- small prep kernels ----------------
__global__ __launch_bounds__(256) void k_escale(const float* __restrict__ He,
                                                const float* __restrict__ p,
                                                float* __restrict__ esc) {
  int e = blockIdx.x * 256 + threadIdx.x;
  if (e >= EE) return;
  const float* row = He + (size_t)e * INE;
  float s = 0.f;
#pragma unroll 8
  for (int k = 0; k < INE; ++k) s += row[k] * p[k];
  esc[e] = s;
}

// Tsc = bf16(T * esc[col]), Tb = bf16(T)   -- single pass over T
__global__ __launch_bounds__(256) void k_Tconv(const float* __restrict__ T,
                                               const float* __restrict__ esc,
                                               __bf16* __restrict__ Tsc,
                                               __bf16* __restrict__ Tb) {
  int k = blockIdx.x * 256 + threadIdx.x;
  int i = blockIdx.y;
  size_t idx = (size_t)i * EE + k;
  float v = T[idx];
  Tb[idx]  = (__bf16)v;
  Tsc[idx] = (__bf16)(v * esc[k]);
}

// dst[c*R + r] = bf16(src[r*C + c])  (convert + transpose, 32x32 LDS tiles)
__global__ __launch_bounds__(256) void k_transpose_bf16(const float* __restrict__ src,
                                                        __bf16* __restrict__ dst,
                                                        int R, int C) {
  __shared__ float tile[32][33];
  int tx = threadIdx.x & 31, ty = threadIdx.x >> 5;
  int rb = blockIdx.y * 32, cb = blockIdx.x * 32;
#pragma unroll
  for (int z = 0; z < 4; ++z)
    tile[ty + z * 8][tx] = src[(size_t)(rb + ty + z * 8) * C + cb + tx];
  __syncthreads();
#pragma unroll
  for (int z = 0; z < 4; ++z)
    dst[(size_t)(cb + ty + z * 8) * R + rb + tx] = (__bf16)tile[tx][ty + z * 8];
}

// ------- kernel: Bmat = 0.5*(M1+1) o adj_v, M1 = (T*esc)@T^T (diag->1) -------
// Block tile 256x128, 8 waves @ 64x64 each: 16 WMMA per K-step per wave.
__global__ __launch_bounds__(256) void k_adj_build(const __bf16* __restrict__ Asc,
                                                   const __bf16* __restrict__ Braw,
                                                   const float* __restrict__ adjv,
                                                   __bf16* __restrict__ Bmat) {
  __shared__ __bf16 As[2][256 * LDT];
  __shared__ __bf16 Bs[2][128 * LDT];

  const int tid  = threadIdx.x;
  const int lane = tid & 31;
  const int wave = tid >> 5;
  const int waveM = (wave >> 1) * 64;   // 0,64,128,192
  const int waveN = (wave & 1) * 64;    // 0,64
  const int jBase = blockIdx.x * 128;
  const int iBase = blockIdx.y * 256;

  const int lr = tid >> 2;              // 0..63
  const int lc = (tid & 3) * 8;         // 0,8,16,24

  f32x8 acc[4][4] = {};

#if USE_ASYNC
  auto issue = [&](int buf, int k0) {
#pragma unroll
    for (int z = 0; z < 4; ++z)
      cp_async16(Asc + (size_t)(iBase + lr + z * 64) * EE + k0 + lc,
                 &As[buf][(lr + z * 64) * LDT + lc]);
#pragma unroll
    for (int z = 0; z < 2; ++z)
      cp_async16(Braw + (size_t)(jBase + lr + z * 64) * EE + k0 + lc,
                 &Bs[buf][(lr + z * 64) * LDT + lc]);
  };
  issue(0, 0);
#endif

  const int NT = EE / TILE_K;
  for (int t = 0; t < NT; ++t) {
    const int p = t & 1;
#if USE_ASYNC
    if (t + 1 < NT) { issue(p ^ 1, (t + 1) * TILE_K); WAIT_ASYNC(6); }
    else            { WAIT_ASYNC(0); }
#else
    {
      const int k0 = t * TILE_K;
#pragma unroll
      for (int z = 0; z < 4; ++z)
        *(bf16x8*)&As[p][(lr + z * 64) * LDT + lc] =
            *(const bf16x8*)(Asc + (size_t)(iBase + lr + z * 64) * EE + k0 + lc);
#pragma unroll
      for (int z = 0; z < 2; ++z)
        *(bf16x8*)&Bs[p][(lr + z * 64) * LDT + lc] =
            *(const bf16x8*)(Braw + (size_t)(jBase + lr + z * 64) * EE + k0 + lc);
    }
#endif
    __syncthreads();

    bf16x16 af[4], bb[4];
#pragma unroll
    for (int a = 0; a < 4; ++a) af[a] = load_a_frag(&As[p][0], waveM + a * 16, lane);
#pragma unroll
    for (int b = 0; b < 4; ++b) bb[b] = load_b_frag(&Bs[p][0], waveN + b * 16, lane);
#pragma unroll
    for (int a = 0; a < 4; ++a)
#pragma unroll
      for (int b = 0; b < 4; ++b)
        acc[a][b] = wmma_bf16(af[a], bb[b], acc[a][b]);
    __syncthreads();
  }

  // epilogue: diag->1, then 0.5*(M1+1)*adj_v, store bf16
  const int h = (lane >> 4) & 1;
  const int n = lane & 15;
#pragma unroll
  for (int a = 0; a < 4; ++a)
#pragma unroll
    for (int b = 0; b < 4; ++b) {
      int j = jBase + waveN + b * 16 + n;
#pragma unroll
      for (int r = 0; r < 8; ++r) {
        int i = iBase + waveM + a * 16 + r + h * 8;
        float v = acc[a][b][r];
        if (i == j) v = 1.0f;
        v = 0.5f * (v + 1.0f) * adjv[(size_t)i * NN + j];
        Bmat[(size_t)i * NN + j] = (__bf16)v;
      }
    }
}

// ------- generic bf16 GEMM: out[M,Nc] = A[M,K] @ BT[Nc,K]^T -------
// A row-major bf16 (stride K); BT is the pre-transposed B (row n = column n),
// so both tiles stage as contiguous row copies. Block 128x128, waves @ 32x64.
template <bool STORE_F32>
__global__ __launch_bounds__(256) void k_gemm(const __bf16* __restrict__ A,
                                              const __bf16* __restrict__ BT,
                                              const float* __restrict__ bias,
                                              void* __restrict__ Cout,
                                              int K, int Nc) {
  __shared__ __bf16 As[2][128 * LDT];
  __shared__ __bf16 Bs[2][128 * LDT];

  const int tid  = threadIdx.x;
  const int lane = tid & 31;
  const int wave = tid >> 5;
  const int waveM = (wave >> 1) * 32;
  const int waveN = (wave & 1) * 64;
  const int nBase = blockIdx.x * 128;
  const int iBase = blockIdx.y * 128;

  const int lr = tid >> 2;              // 0..63
  const int lc = (tid & 3) * 8;         // 0,8,16,24

  f32x8 acc[2][4] = {};

#if USE_ASYNC
  auto issue = [&](int buf, int k0) {
#pragma unroll
    for (int z = 0; z < 2; ++z)
      cp_async16(A + (size_t)(iBase + lr + z * 64) * K + k0 + lc,
                 &As[buf][(lr + z * 64) * LDT + lc]);
#pragma unroll
    for (int z = 0; z < 2; ++z)
      cp_async16(BT + (size_t)(nBase + lr + z * 64) * K + k0 + lc,
                 &Bs[buf][(lr + z * 64) * LDT + lc]);
  };
  issue(0, 0);
#endif

  const int NT = K / TILE_K;
  for (int t = 0; t < NT; ++t) {
    const int p = t & 1;
#if USE_ASYNC
    if (t + 1 < NT) { issue(p ^ 1, (t + 1) * TILE_K); WAIT_ASYNC(4); }
    else            { WAIT_ASYNC(0); }
#else
    {
      const int k0 = t * TILE_K;
#pragma unroll
      for (int z = 0; z < 2; ++z)
        *(bf16x8*)&As[p][(lr + z * 64) * LDT + lc] =
            *(const bf16x8*)(A + (size_t)(iBase + lr + z * 64) * K + k0 + lc);
#pragma unroll
      for (int z = 0; z < 2; ++z)
        *(bf16x8*)&Bs[p][(lr + z * 64) * LDT + lc] =
            *(const bf16x8*)(BT + (size_t)(nBase + lr + z * 64) * K + k0 + lc);
    }
#endif
    __syncthreads();

    bf16x16 af[2], bb[4];
#pragma unroll
    for (int a = 0; a < 2; ++a) af[a] = load_a_frag(&As[p][0], waveM + a * 16, lane);
#pragma unroll
    for (int b = 0; b < 4; ++b) bb[b] = load_b_frag(&Bs[p][0], waveN + b * 16, lane);
#pragma unroll
    for (int a = 0; a < 2; ++a)
#pragma unroll
      for (int b = 0; b < 4; ++b)
        acc[a][b] = wmma_bf16(af[a], bb[b], acc[a][b]);
    __syncthreads();
  }

  const int h = (lane >> 4) & 1;
  const int n = lane & 15;
#pragma unroll
  for (int a = 0; a < 2; ++a)
#pragma unroll
    for (int b = 0; b < 4; ++b) {
      int j = nBase + waveN + b * 16 + n;
#pragma unroll
      for (int r = 0; r < 8; ++r) {
        int i = iBase + waveM + a * 16 + r + h * 8;
        float v = acc[a][b][r];
        if (STORE_F32) {
          ((float*)Cout)[(size_t)i * Nc + j] = v + bias[j];
        } else {
          ((__bf16*)Cout)[(size_t)i * Nc + j] = (__bf16)v;
        }
      }
    }
}

// ---------------- launcher ----------------
extern "C" void kernel_launch(void* const* d_in, const int* in_sizes, int n_in,
                              void* d_out, int out_size, void* d_ws, size_t ws_size,
                              hipStream_t stream) {
  const float* H_v    = (const float*)d_in[0];
  const float* H_e    = (const float*)d_in[1];
  // d_in[2] = adj_e (unused on the node_layer path)
  const float* adj_v  = (const float*)d_in[3];
  const float* T      = (const float*)d_in[4];
  const float* weight = (const float*)d_in[5];
  const float* p      = (const float*)d_in[6];
  const float* bias   = (const float*)d_in[7];

  char* ws = (char*)d_ws;
  float*  esc    = (float*)(ws + 0);                 //  32 KB
  __bf16* Tsc_bf = (__bf16*)(ws + 32768);            //  64 MB  bf16(T*esc)
  __bf16* T_bf   = (__bf16*)(ws + 67141632);         //  64 MB  bf16(T)
  __bf16* HvT_bf = (__bf16*)(ws + 134250496);        //   4 MB  bf16(H_v^T) [512,4096]
  __bf16* WT_bf  = (__bf16*)(ws + 138444800);        // 512 KB  bf16(W^T)   [512,512]
  __bf16* C2_bf  = (__bf16*)(ws + 138969088);        //   4 MB
  __bf16* Bmat   = (__bf16*)(ws + 143163392);        //  32 MB

  float* ret    = (float*)d_out;                     // [4096,512]
  float* he_out = ret + (size_t)NN * OUTV;           // [8192,128]

  // 1) edge scale vector
  k_escale<<<EE / 256, 256, 0, stream>>>(H_e, p, esc);

  // 2) bf16 precondition passes (bandwidth-trivial vs compute)
  k_Tconv<<<dim3(EE / 256, NN), 256, 0, stream>>>(T, esc, Tsc_bf, T_bf);
  k_transpose_bf16<<<dim3(INV / 32, NN / 32), 256, 0, stream>>>(H_v, HvT_bf, NN, INV);
  k_transpose_bf16<<<dim3(OUTV / 32, INV / 32), 256, 0, stream>>>(weight, WT_bf, INV, OUTV);

  // 3) the 275-GFLOP GEMM with fused adjacency epilogue
  k_adj_build<<<dim3(NN / 128, NN / 256), 256, 0, stream>>>(Tsc_bf, T_bf, adj_v, Bmat);

  // 4) C2 = Bmat @ H_v
  k_gemm<false><<<dim3(INV / 128, NN / 128), 256, 0, stream>>>(
      Bmat, HvT_bf, nullptr, C2_bf, NN, INV);

  // 5) ret = C2 @ W + bias
  k_gemm<true><<<dim3(OUTV / 128, NN / 128), 256, 0, stream>>>(
      C2_bf, WT_bf, bias, ret, INV, OUTV);

  // 6) second tuple element: H_e passthrough
  (void)hipMemcpyAsync(he_out, H_e, (size_t)EE * INE * sizeof(float),
                       hipMemcpyDeviceToDevice, stream);
}
